// MultiHeadAttention_1159641170400
// MI455X (gfx1250) — matrix operand order repaired and checked by reading, not compiled, not run on hardware
//
#include <hip/hip_runtime.h>
#include <hip/hip_bf16.h>

typedef __attribute__((ext_vector_type(16))) _Float16 v16h;
typedef __attribute__((ext_vector_type(8)))  _Float16 v8h;
typedef __attribute__((ext_vector_type(8)))  float    v8f;

// Types for the async-to-LDS builtin (per hipcc diagnostic: param is
// 'int __vector(4) __device__ *').
typedef int v4i_ __attribute__((__vector_size__(16)));
typedef __attribute__((address_space(1))) v4i_* gas_p;
typedef __attribute__((address_space(3))) v4i_* las_p;

#define B_  4
#define S_  2048
#define H_  16
#define HD_ 64
#define D_  1024

#if __has_builtin(__builtin_amdgcn_global_load_async_to_lds_b128)
#define HAVE_ASYNC_LDS 1
#else
#define HAVE_ASYNC_LDS 0
#endif

__device__ __forceinline__ v8f wmma_f16(v16h a, v16h b, v8f c) {
  // D = A(16x32 f16) * B(32x16 f16) + C(16x16 f32)
  return __builtin_amdgcn_wmma_f32_16x16x32_f16(false, a, false, b, (short)0, c,
                                                false, false);
}

// A-fragment: lane l holds row (l&15); 16 contiguous halves at K = (l>>4)*16..+15
__device__ __forceinline__ v16h frag_a(const _Float16* rowbase, int lhi) {
  const v8h lo = *(const v8h*)(rowbase + lhi * 16);
  const v8h hi = *(const v8h*)(rowbase + lhi * 16 + 8);
  return __builtin_shufflevector(lo, hi, 0, 1, 2, 3, 4, 5, 6, 7,
                                 8, 9, 10, 11, 12, 13, 14, 15);
}

// B-fragment from N-major (NxK) storage: lane l holds col N=(l&15),
// 16 contiguous halves at K = (l>>4)*16. Caller passes fully-resolved pointer.
__device__ __forceinline__ v16h frag_b(const _Float16* p) {
  const v8h lo = *(const v8h*)(p);
  const v8h hi = *(const v8h*)(p + 8);
  return __builtin_shufflevector(lo, hi, 0, 1, 2, 3, 4, 5, 6, 7,
                                 8, 9, 10, 11, 12, 13, 14, 15);
}

// Convert 8 f32 -> 8 f16 with v_cvt_pk_rtz_f16_f32 and store 16B to LDS.
__device__ __forceinline__ void cvt_store8(_Float16* dst, float4 a, float4 b) {
  auto p0 = __builtin_amdgcn_cvt_pkrtz(a.x, a.y);
  auto p1 = __builtin_amdgcn_cvt_pkrtz(a.z, a.w);
  auto p2 = __builtin_amdgcn_cvt_pkrtz(b.x, b.y);
  auto p3 = __builtin_amdgcn_cvt_pkrtz(b.z, b.w);
  auto q0 = __builtin_shufflevector(p0, p1, 0, 1, 2, 3);
  auto q1 = __builtin_shufflevector(p2, p3, 0, 1, 2, 3);
  auto q  = __builtin_shufflevector(q0, q1, 0, 1, 2, 3, 4, 5, 6, 7);
  *(__typeof__(q)*)dst = q;
}

// ---------------------------------------------------------------------------
// GEMM: Y[M,N] = A[M,K] * Bt[N,K]^T   (K = N = 1024, M = 8192)
// MODE 0: Y -> Qh f16 [B,H,S,HD];  MODE 1: Y -> Kh f16 [B,H,S,HD]
// MODE 2: Y -> Vt f16 [B,H,HD,S];  MODE 3: Y -> out f32 + bias (A is f16 ctx)
// 256 thr (8 waves), block tile 128x128, staged K-tile 64 -> 16 WMMA/barrier.
// ---------------------------------------------------------------------------
template <int MODE>
__global__ __launch_bounds__(256) void gemm_wmma_k(const void* __restrict__ Ap,
                                                   const float* __restrict__ Bt,
                                                   const float* __restrict__ bias,
                                                   _Float16* __restrict__ out16,
                                                   float* __restrict__ out32) {
  __shared__ _Float16 As[128 * 72];  // 128 x 64, pitch 72 halves (144B)
  __shared__ _Float16 Bs[128 * 72];

  const int tid = threadIdx.x;
  const int wave = tid >> 5, lane = tid & 31;
  const int lhi = lane >> 4, llo = lane & 15;
  const int bM = blockIdx.x * 128, bN = blockIdx.y * 128;
  const int wM = (wave >> 2) * 64, wN = (wave & 3) * 32;

  const v8f z = {0.f, 0.f, 0.f, 0.f, 0.f, 0.f, 0.f, 0.f};
  v8f acc[4][2];
#pragma unroll
  for (int mi = 0; mi < 4; ++mi)
#pragma unroll
    for (int ni = 0; ni < 2; ++ni) acc[mi][ni] = z;

  const int row = tid >> 1, off = (tid & 1) * 32;  // 2 thr/row, 32 halves each

  for (int k0 = 0; k0 < D_; k0 += 64) {
    if (MODE < 3) {
      const float* ap = (const float*)Ap + (size_t)(bM + row) * D_ + k0 + off;
      if (k0 + 64 < D_) __builtin_prefetch(ap + 64, 0, 0);
      const float4* af = (const float4*)ap;
      cvt_store8(As + row * 72 + off,      af[0], af[1]);
      cvt_store8(As + row * 72 + off + 8,  af[2], af[3]);
      cvt_store8(As + row * 72 + off + 16, af[4], af[5]);
      cvt_store8(As + row * 72 + off + 24, af[6], af[7]);
    } else {
      const _Float16* ap = (const _Float16*)Ap + (size_t)(bM + row) * D_ + k0 + off;
      if (k0 + 64 < D_) __builtin_prefetch(ap + 64, 0, 0);
      const uint4* af = (const uint4*)ap;
      uint4* d = (uint4*)(As + row * 72 + off);
      d[0] = af[0]; d[1] = af[1]; d[2] = af[2]; d[3] = af[3];
    }
    {
      const float* bp = Bt + (size_t)(bN + row) * D_ + k0 + off;
      if (k0 + 64 < D_) __builtin_prefetch(bp + 64, 0, 0);
      const float4* bf = (const float4*)bp;
      cvt_store8(Bs + row * 72 + off,      bf[0], bf[1]);
      cvt_store8(Bs + row * 72 + off + 8,  bf[2], bf[3]);
      cvt_store8(Bs + row * 72 + off + 16, bf[4], bf[5]);
      cvt_store8(Bs + row * 72 + off + 24, bf[6], bf[7]);
    }
    __syncthreads();

#pragma unroll
    for (int kk = 0; kk < 2; ++kk) {
      v16h aF[4], bF[2];
#pragma unroll
      for (int mi = 0; mi < 4; ++mi)
        aF[mi] = frag_a(As + (wM + mi * 16 + llo) * 72 + kk * 32, lhi);
#pragma unroll
      for (int ni = 0; ni < 2; ++ni)
        bF[ni] = frag_b(Bs + (wN + ni * 16 + llo) * 72 + kk * 32 + lhi * 16);
#pragma unroll
      for (int mi = 0; mi < 4; ++mi)
#pragma unroll
        for (int ni = 0; ni < 2; ++ni)
          acc[mi][ni] = wmma_f16(aF[mi], bF[ni], acc[mi][ni]);
    }
    __syncthreads();
  }

  // Epilogue: C element r <-> row r + 8*(lane>>4), col (lane&15)
#pragma unroll
  for (int mi = 0; mi < 4; ++mi)
#pragma unroll
    for (int ni = 0; ni < 2; ++ni)
#pragma unroll
      for (int r = 0; r < 8; ++r) {
        const int grow = bM + wM + mi * 16 + r + 8 * lhi;  // token n = b*S + s
        const int gcol = bN + wN + ni * 16 + llo;          // feature e = h*HD + d
        const float v = acc[mi][ni][r];
        if (MODE == 0 || MODE == 1) {
          const int b = grow >> 11, s = grow & (S_ - 1);
          const int h = gcol >> 6, d = gcol & (HD_ - 1);
          out16[(((size_t)b * H_ + h) * S_ + s) * HD_ + d] = (_Float16)v;
        } else if (MODE == 2) {
          const int b = grow >> 11, s = grow & (S_ - 1);
          const int h = gcol >> 6, d = gcol & (HD_ - 1);
          out16[(((size_t)b * H_ + h) * HD_ + d) * S_ + s] = (_Float16)v;
        } else {
          out32[(size_t)grow * D_ + gcol] = v + bias[gcol];
        }
      }
}

// ---------------------------------------------------------------------------
// Flash attention, causal. Block = (128 queries) x one (b,h). 8 waves,
// each wave owns 16 query rows. Key blocks of Bc=64, exp2-domain softmax.
// Qh,Kh: [B,H,S,64] f16; Vt: [B,H,64,S] f16; Ctx out: [B,S,H*64] f16.
// ---------------------------------------------------------------------------
__global__ __launch_bounds__(256) void flash_attn(const _Float16* __restrict__ Qh,
                                                  const _Float16* __restrict__ Kh,
                                                  const _Float16* __restrict__ Vt,
                                                  _Float16* __restrict__ Ctx) {
  __shared__ _Float16 Qs[128 * 72];     // 128 x 64, pitch 72
  __shared__ _Float16 Ks[64 * 72];      // 64 keys x 64 hd
  __shared__ _Float16 Vts[64 * 72];     // 64 hd  x 64 keys (N x K for PV)
  __shared__ _Float16 Ps[8 * 16 * 72];  // per-wave P staging (16 x 64)

  const int qt = blockIdx.x;  // query tile: rows [qt*128, qt*128+128)
  const int bh = blockIdx.y;  // b*H + h
  const int tid = threadIdx.x;
  const int wave = tid >> 5, lane = tid & 31;
  const int lhi = lane >> 4, llo = lane & 15;

  const _Float16* Qg = Qh + ((size_t)bh * S_ + qt * 128) * HD_;
  const _Float16* Kg = Kh + (size_t)bh * S_ * HD_;
  const _Float16* Vg = Vt + (size_t)bh * HD_ * S_;

  // Stage Q tile (stays resident): 128 rows x 64 halves
  {
    const int row = tid >> 1, off = (tid & 1) * 32;
#if HAVE_ASYNC_LDS
#pragma unroll
    for (int j = 0; j < 4; ++j)
      __builtin_amdgcn_global_load_async_to_lds_b128(
          (gas_p)(Qg + row * HD_ + off + j * 8),
          (las_p)(Qs + row * 72 + off + j * 8), 0, 0);
#else
    const uint4* src = (const uint4*)(Qg + row * HD_ + off);
    uint4* dst = (uint4*)(Qs + row * 72 + off);
    dst[0] = src[0]; dst[1] = src[1]; dst[2] = src[2]; dst[3] = src[3];
#endif
  }

  const v8f z = {0.f, 0.f, 0.f, 0.f, 0.f, 0.f, 0.f, 0.f};
  v8f o[4];
#pragma unroll
  for (int n4 = 0; n4 < 4; ++n4) o[n4] = z;
  float rm[8], rs[8];
#pragma unroll
  for (int r = 0; r < 8; ++r) { rm[r] = -1e30f; rs[r] = 0.f; }

  // scores scaled by 1/sqrt(64) and converted to log2 domain in one constant
  const float SC = 0.125f * 1.44269504088896340736f;

  const int kbMax = 2 * qt + 1;  // keys up to 128*(qt+1) cover the diagonal
  for (int kb = 0; kb <= kbMax; ++kb) {
    __syncthreads();  // protect K/V/P LDS reuse (covers Q staging on iter 0)
    {
      const int row = tid >> 2, off = (tid & 3) * 16;  // 4 thr/row
#if HAVE_ASYNC_LDS
#pragma unroll
      for (int j = 0; j < 2; ++j) {
        __builtin_amdgcn_global_load_async_to_lds_b128(
            (gas_p)(Kg + (size_t)(kb * 64 + row) * HD_ + off + j * 8),
            (las_p)(Ks + row * 72 + off + j * 8), 0, 0);
        __builtin_amdgcn_global_load_async_to_lds_b128(
            (gas_p)(Vg + (size_t)row * S_ + kb * 64 + off + j * 8),
            (las_p)(Vts + row * 72 + off + j * 8), 0, 0);
      }
      asm volatile("s_wait_asynccnt 0" ::: "memory");
#else
      const uint4* ks = (const uint4*)(Kg + (size_t)(kb * 64 + row) * HD_ + off);
      uint4* kd = (uint4*)(Ks + row * 72 + off);
      kd[0] = ks[0]; kd[1] = ks[1];
      const uint4* vs = (const uint4*)(Vg + (size_t)row * S_ + kb * 64 + off);
      uint4* vd = (uint4*)(Vts + row * 72 + off);
      vd[0] = vs[0]; vd[1] = vs[1];
#endif
    }
    __syncthreads();

    // ---- S = Q K^T for this wave's 16 rows x 64 keys ----
    v8f s[4];
#pragma unroll
    for (int ni = 0; ni < 4; ++ni) {
      v8f c = z;
#pragma unroll
      for (int kk = 0; kk < 2; ++kk) {  // hd = 2 chunks of 32
        v16h a = frag_a(Qs + (wave * 16 + llo) * 72 + kk * 32, lhi);
        v16h b = frag_b(Ks + (ni * 16 + llo) * 72 + kk * 32 + lhi * 16);
        c = wmma_f16(a, b, c);
      }
      s[ni] = c;
    }

    // ---- scale (+ causal mask only when this wave touches the diagonal) ----
    const bool needMask = (kb * 64 + 63) > (qt * 128 + wave * 16);  // wave-uniform
    if (needMask) {
#pragma unroll
      for (int ni = 0; ni < 4; ++ni)
#pragma unroll
        for (int r = 0; r < 8; ++r) {
          const int kcol = kb * 64 + ni * 16 + llo;
          const int qrow = qt * 128 + wave * 16 + r + 8 * lhi;
          const float v = s[ni][r] * SC;
          s[ni][r] = (kcol > qrow) ? -1e30f : v;
        }
    } else {
#pragma unroll
      for (int ni = 0; ni < 4; ++ni)
#pragma unroll
        for (int r = 0; r < 8; ++r) s[ni][r] *= SC;
    }

    // ---- online softmax in exp2 domain (row = lanes sharing llo group) ----
    float corr[8];
#pragma unroll
    for (int r = 0; r < 8; ++r) {
      float m = fmaxf(fmaxf(s[0][r], s[1][r]), fmaxf(s[2][r], s[3][r]));
      m = fmaxf(m, __shfl_xor(m, 1, 32));
      m = fmaxf(m, __shfl_xor(m, 2, 32));
      m = fmaxf(m, __shfl_xor(m, 4, 32));
      m = fmaxf(m, __shfl_xor(m, 8, 32));
      const float mn = fmaxf(rm[r], m);
      corr[r] = exp2f(rm[r] - mn);
      rm[r] = mn;
      float psum = 0.f;
#pragma unroll
      for (int ni = 0; ni < 4; ++ni) {
        const float p = exp2f(s[ni][r] - mn);
        s[ni][r] = p;
        psum += p;
      }
      psum += __shfl_xor(psum, 1, 32);
      psum += __shfl_xor(psum, 2, 32);
      psum += __shfl_xor(psum, 4, 32);
      psum += __shfl_xor(psum, 8, 32);
      rs[r] = rs[r] * corr[r] + psum;
    }

    // ---- re-shape P (C-layout) -> A-layout via per-wave LDS ----
    _Float16* pw = Ps + wave * 16 * 72;
#pragma unroll
    for (int ni = 0; ni < 4; ++ni)
#pragma unroll
      for (int r = 0; r < 8; ++r)
        pw[(r + 8 * lhi) * 72 + ni * 16 + llo] = (_Float16)s[ni][r];
#pragma unroll
    for (int n4 = 0; n4 < 4; ++n4)
#pragma unroll
      for (int r = 0; r < 8; ++r) o[n4][r] *= corr[r];
    asm volatile("s_wait_dscnt 0" ::: "memory");  // wave-local LDS RAW fence

    // ---- O += P V  (B-operand = Vts is hd x keys, i.e. N x K) ----
#pragma unroll
    for (int kc = 0; kc < 2; ++kc) {  // 64 keys = 2 chunks of 32
      v16h a = frag_a(pw + llo * 72 + kc * 32, lhi);
#pragma unroll
      for (int n4 = 0; n4 < 4; ++n4) {
        v16h b = frag_b(Vts + (n4 * 16 + llo) * 72 + kc * 32 + lhi * 16);
        o[n4] = wmma_f16(a, b, o[n4]);
      }
    }
  }

  // ---- normalize + write ctx f16 [B, S, H*64] ----
  const int b = bh >> 4, h = bh & 15;
#pragma unroll
  for (int n4 = 0; n4 < 4; ++n4)
#pragma unroll
    for (int r = 0; r < 8; ++r) {
      const int qrow = qt * 128 + wave * 16 + r + 8 * lhi;
      const float v = o[n4][r] / rs[r];
      Ctx[((size_t)b * S_ + qrow) * D_ + h * HD_ + n4 * 16 + llo] = (_Float16)v;
    }
}

// ---------------------------------------------------------------------------
extern "C" void kernel_launch(void* const* d_in, const int* in_sizes, int n_in,
                              void* d_out, int out_size, void* d_ws, size_t ws_size,
                              hipStream_t stream) {
  (void)in_sizes; (void)n_in; (void)out_size; (void)ws_size;
  const float* x  = (const float*)d_in[0];
  const float* Wq = (const float*)d_in[1];
  const float* Wk = (const float*)d_in[2];
  const float* Wv = (const float*)d_in[3];
  const float* Wo = (const float*)d_in[4];
  const float* bo = (const float*)d_in[5];
  float* out = (float*)d_out;

  const size_t NE = (size_t)B_ * S_ * D_;  // 8,388,608 elements
  _Float16* Qh  = (_Float16*)d_ws;         // 16 MB each
  _Float16* Kh  = Qh + NE;
  _Float16* Vt  = Kh + NE;
  _Float16* Ctx = Vt + NE;

  const dim3 blk(256);
  const dim3 gGemm(64, 8);                 // 8192/128 x 1024/128
  gemm_wmma_k<0><<<gGemm, blk, 0, stream>>>((const void*)x, Wq, nullptr, Qh, nullptr);
  gemm_wmma_k<1><<<gGemm, blk, 0, stream>>>((const void*)x, Wk, nullptr, Kh, nullptr);
  gemm_wmma_k<2><<<gGemm, blk, 0, stream>>>((const void*)x, Wv, nullptr, Vt, nullptr);
  flash_attn<<<dim3(16, 64), blk, 0, stream>>>(Qh, Kh, Vt, Ctx);
  gemm_wmma_k<3><<<gGemm, blk, 0, stream>>>((const void*)Ctx, Wo, bo, nullptr, out);
}
